// Net_77446850281992
// MI455X (gfx1250) — compile-verified, matrix operand order checked by reading, not run on hardware
//
#include <hip/hip_runtime.h>
#include <math.h>

// Problem constants (match reference)
#define BB 256
#define VV 128
#define MM 256
#define HH 1024
#define TDIM 10
#define IN_DIM 35840           // (V + TDIM + 1 + 1) * M
#define KSPLIT 4
#define KPER (IN_DIM / KSPLIT) // 8960
#define KC 64
#define NITER (KPER / KC)      // 140

#define LDA 72                 // padded LDS row stride (bf16 elements)
#define A_TILE (BB * LDA)      // 18432 ushorts = 36,864 B
#define W_TILE (16 * LDA)      //  1152 ushorts =  2,304 B
#define SMEM_BYTES ((2 * A_TILE + 2 * W_TILE) * 2)

typedef __attribute__((ext_vector_type(16))) __bf16 v16bf;
typedef __attribute__((ext_vector_type(8)))  float  v8f;
typedef __attribute__((ext_vector_type(4)))  unsigned v4u;
typedef __attribute__((ext_vector_type(8)))  unsigned v8u;

union BfOp { v16bf v; uint4 q[2]; };

__device__ __forceinline__ unsigned short f2bf(float f) {
  unsigned u = __float_as_uint(f);
  u += 0x7FFFu + ((u >> 16) & 1u);   // round-to-nearest-even
  return (unsigned short)(u >> 16);
}

// ---------------------------------------------------------------------------
// Tensor Data Mover: DMA a 256-row x 64-col bf16 tile (row stride IN_DIM)
// from global into LDS, inserting 4 DWORDs of padding every 32 DWORDs so the
// LDS tile lands with a 72-element row stride (bank-conflict padding in HW).
// D# layout per CDNA5 ISA (group0 + group1, 2D tile).
// ---------------------------------------------------------------------------
__device__ __forceinline__ void tdm_load_A_tile(unsigned long long gaddr,
                                                unsigned ldsaddr) {
  v4u g0;
  g0[0] = 1u;                                   // count=1, user mode
  g0[1] = ldsaddr;                              // lds_addr (bytes)
  g0[2] = (unsigned)gaddr;                      // global_addr[31:0]
  g0[3] = (unsigned)(gaddr >> 32) | (2u << 30); // global_addr[56:32] | type=2
  v8u g1;
  g1[0] = (1u << 16)      // data_size = 1 (2 bytes)
        | (1u << 20)      // pad_enable
        | (4u << 22)      // pad_interval = 32 DWORDs (one 64-elem bf16 row)
        | (3u << 25);     // pad_amount   = 4 DWORDs (8 bf16)
  g1[1] = ((unsigned)IN_DIM & 0xFFFFu) << 16;   // tensor_dim0[15:0] @ bits[79:48]
  g1[2] = ((unsigned)MM & 0xFFFFu) << 16;       // tensor_dim0 hi=0 | tensor_dim1 lo
  g1[3] = ((unsigned)KC) << 16;                 // tensor_dim1 hi=0 | tile_dim0=64
  g1[4] = (unsigned)MM;                         // tile_dim1=256 | tile_dim2=0
  g1[5] = (unsigned)IN_DIM;                     // tensor_dim0_stride[31:0]
  g1[6] = 0u;                                   // stride hi | dim1_stride lo
  g1[7] = 0u;
  asm volatile("tensor_load_to_lds %0, %1" :: "s"(g0), "s"(g1) : "memory");
}

// ---------------------------------------------------------------------------
// Kernel 1: surprise + ring-roll + stable sort + binary/norm encode -> bf16 A
// One block per batch row. key = (timing<<8)|slot reproduces stable argsort.
// ---------------------------------------------------------------------------
__global__ void prep_kernel(const float* __restrict__ x,
                            const float* __restrict__ memory,
                            const int*   __restrict__ mtim,
                            const float* __restrict__ msur,
                            const float* __restrict__ lastp,
                            unsigned short* __restrict__ Abf)
{
  __shared__ unsigned skey[MM];
  __shared__ float    ssurp[MM];
  __shared__ float    red[128];
  __shared__ float    sbc;

  const int b = blockIdx.x;
  const int t = threadIdx.x;

  if (t < 128) red[t] = x[b * VV + t] * lastp[b * VV + t];
  __syncthreads();
  for (int s = 64; s > 0; s >>= 1) {
    if (t < s) red[t] += red[t + s];
    __syncthreads();
  }
  if (t == 0) sbc = -logf(red[0] + 1e-8f);
  __syncthreads();
  const float surprise = sbc;

  {
    int timing = (t == 0) ? 0 : (mtim[b * MM + t - 1] + 1);
    ssurp[t]   = (t == 0) ? surprise : msur[b * MM + t - 1] * 0.99f;
    skey[t]    = ((unsigned)timing << 8) | (unsigned)t;
  }
  __syncthreads();

  for (int k = 2; k <= MM; k <<= 1) {
    for (int j = k >> 1; j > 0; j >>= 1) {
      int ixj = t ^ j;
      if (ixj > t) {
        unsigned a = skey[t], c = skey[ixj];
        bool up = ((t & k) == 0);
        if ((a > c) == up) { skey[t] = c; skey[ixj] = a; }
      }
      __syncthreads();
    }
  }

  const unsigned maxT = skey[MM - 1] >> 8;
  const size_t rowbase = (size_t)b * IN_DIM;

  for (int pair = 0; pair < MM / 2; ++pair) {
    int mm  = pair * 2 + (t >> 7);
    int v   = t & 127;
    int idx = (int)(skey[mm] & 255u);
    float val = (idx == 0) ? x[b * VV + v]
                           : memory[((size_t)b * MM + (idx - 1)) * VV + v];
    Abf[rowbase + (size_t)mm * VV + v] = f2bf(val);
  }

  {
    unsigned key = skey[t];
    int st  = (int)(key >> 8);
    int idx = (int)(key & 255u);
    size_t o = rowbase + MM * VV;
    #pragma unroll
    for (int i = 0; i < TDIM; ++i)
      Abf[o + (size_t)t * TDIM + i] = f2bf((float)((st >> i) & 1));
    Abf[rowbase + MM * VV + MM * TDIM + t]      = f2bf((float)st / ((float)maxT + 1.0f));
    Abf[rowbase + MM * VV + MM * TDIM + MM + t] = f2bf(ssurp[idx]);
  }
}

// ---------------------------------------------------------------------------
// Kernel 2: W2 [1024x128] fp32 -> W2T [128x1024] bf16
// ---------------------------------------------------------------------------
__global__ void w2t_kernel(const float* __restrict__ W2,
                           unsigned short* __restrict__ W2T)
{
  int id = blockIdx.x * blockDim.x + threadIdx.x;
  int k = id >> 7, n = id & 127;
  W2T[(size_t)n * HH + k] = f2bf(W2[id]);
}

// ---------------------------------------------------------------------------
// Kernel 3: dual GEMM, TDM-staged activations, double-buffered LDS.
// Per iter: wave0 issues TDM for tile kk+1; all threads issue W-loads for
// kk+1; compute WMMAs on tile kk; convert/store W; wave0 drains TENSORcnt;
// one barrier publishes both buffers.
// ---------------------------------------------------------------------------
__global__ void gemm1_kernel(const unsigned short* __restrict__ Abf,
                             const float* __restrict__ W1,
                             const float* __restrict__ Wg,
                             float* __restrict__ partial)
{
  extern __shared__ __align__(16) unsigned short smem[];
  unsigned short* bufA0 = smem;
  unsigned short* bufA1 = smem + A_TILE;
  unsigned short* bufW0 = smem + 2 * A_TILE;
  unsigned short* bufW1 = smem + 2 * A_TILE + W_TILE;

  const int t    = threadIdx.x;
  const int n0   = blockIdx.x * 16;
  const int mat  = blockIdx.y;
  const int z    = blockIdx.z;
  const float* W = mat ? Wg : W1;
  const int kbase = z * KPER;

  const int wave  = t >> 5, lane = t & 31;
  const int lrow  = lane & 15;
  const int khalf = (lane >> 4) << 3;
  const int m0    = wave * 32;
  const int kr    = t >> 2;          // W-tile row this thread loads
  const int c4    = (t & 3) << 2;    // W-tile col group

  const unsigned long long gA = (unsigned long long)(const void*)Abf;
  const unsigned ldsA0 = (unsigned)(unsigned long long)(void*)bufA0;
  const unsigned ldsA1 = (unsigned)(unsigned long long)(void*)bufA1;

  v8f acc0 = {}; v8f acc1 = {};

  // ---- prologue: stage tile 0 ----
  if (wave == 0)
    tdm_load_A_tile(gA + (unsigned long long)(kbase) * 2ull, ldsA0);
  {
    float4 w = *(const float4*)(W + (size_t)(kbase + kr) * HH + n0 + c4);
    bufW0[(c4 + 0) * LDA + kr] = f2bf(w.x);
    bufW0[(c4 + 1) * LDA + kr] = f2bf(w.y);
    bufW0[(c4 + 2) * LDA + kr] = f2bf(w.z);
    bufW0[(c4 + 3) * LDA + kr] = f2bf(w.w);
  }
  if (wave == 0) __builtin_amdgcn_s_wait_tensorcnt(0);
  __syncthreads();

  for (int kk = 0; kk < NITER; ++kk) {
    const int k0 = kbase + kk * KC;
    const unsigned short* lA = (kk & 1) ? bufA1 : bufA0;
    const unsigned short* lW = (kk & 1) ? bufW1 : bufW0;
    unsigned short* lWn      = (kk & 1) ? bufW0 : bufW1;
    const unsigned   ldsAn   = (kk & 1) ? ldsA0 : ldsA1;
    const bool more = (kk + 1 < NITER);

    // issue next tile's DMA + weight loads before computing
    if (more && wave == 0)
      tdm_load_A_tile(gA + (unsigned long long)(k0 + KC) * 2ull, ldsAn);
    float4 wnext;
    if (more)
      wnext = *(const float4*)(W + (size_t)(k0 + KC + kr) * HH + n0 + c4);

    #pragma unroll
    for (int s = 0; s < 2; ++s) {
      const int kb = s * 32 + khalf;
      BfOp bop, a0, a1;
      bop.q[0] = *(const uint4*)(&lW[lrow * LDA + kb]);
      bop.q[1] = *(const uint4*)(&lW[lrow * LDA + kb + 16]);
      a0.q[0]  = *(const uint4*)(&lA[(m0 + lrow) * LDA + kb]);
      a0.q[1]  = *(const uint4*)(&lA[(m0 + lrow) * LDA + kb + 16]);
      a1.q[0]  = *(const uint4*)(&lA[(m0 + 16 + lrow) * LDA + kb]);
      a1.q[1]  = *(const uint4*)(&lA[(m0 + 16 + lrow) * LDA + kb + 16]);
      acc0 = __builtin_amdgcn_wmma_f32_16x16x32_bf16(false, a0.v, false, bop.v,
                                                     (short)0, acc0, false, false);
      acc1 = __builtin_amdgcn_wmma_f32_16x16x32_bf16(false, a1.v, false, bop.v,
                                                     (short)0, acc1, false, false);
    }

    if (more) {
      lWn[(c4 + 0) * LDA + kr] = f2bf(wnext.x);
      lWn[(c4 + 1) * LDA + kr] = f2bf(wnext.y);
      lWn[(c4 + 2) * LDA + kr] = f2bf(wnext.z);
      lWn[(c4 + 3) * LDA + kr] = f2bf(wnext.w);
      if (wave == 0) __builtin_amdgcn_s_wait_tensorcnt(0);
    }
    __syncthreads();
  }

  float* P = partial + ((size_t)z * 2 + mat) * (size_t)(BB * HH);
  const int cn = n0 + lrow;
  const int rbase = m0 + ((lane >> 4) << 3);
  #pragma unroll
  for (int r = 0; r < 8; ++r) {
    P[(size_t)(rbase + r) * HH + cn]      = acc0[r];
    P[(size_t)(rbase + 16 + r) * HH + cn] = acc1[r];
  }
}

// ---------------------------------------------------------------------------
// Kernel 4: reduce K-splits, add biases, apply gate, emit h as bf16
// ---------------------------------------------------------------------------
__global__ void gate_kernel(const float* __restrict__ partial,
                            const float* __restrict__ b1,
                            const float* __restrict__ bg,
                            unsigned short* __restrict__ hbf)
{
  int id = blockIdx.x * blockDim.x + threadIdx.x;
  int j = id & (HH - 1);
  float s1 = 0.f, sg = 0.f;
  #pragma unroll
  for (int z = 0; z < KSPLIT; ++z) {
    s1 += partial[((size_t)z * 2 + 0) * (size_t)(BB * HH) + id];
    sg += partial[((size_t)z * 2 + 1) * (size_t)(BB * HH) + id];
  }
  float g = 1.0f / (1.0f + expf(-(sg + bg[j])));
  hbf[id] = f2bf((s1 + b1[j]) * g);
}

// ---------------------------------------------------------------------------
// Kernel 5: out[256x128] = h @ W2 + b2, WMMA with direct global fragments
// ---------------------------------------------------------------------------
__global__ void gemm2_kernel(const unsigned short* __restrict__ hbf,
                             const unsigned short* __restrict__ W2T,
                             const float* __restrict__ b2,
                             float* __restrict__ out)
{
  const int t = threadIdx.x;
  const int wave = t >> 5, lane = t & 31;
  const int lrow  = lane & 15;
  const int khalf = (lane >> 4) << 3;
  const int row0 = blockIdx.x * 16;
  const int n0   = wave * 16;
  v8f acc = {};
  for (int s = 0; s < HH / 32; ++s) {
    const int kb = s * 32 + khalf;
    BfOp a, bb;
    a.q[0]  = *(const uint4*)(hbf + (size_t)(row0 + lrow) * HH + kb);
    a.q[1]  = *(const uint4*)(hbf + (size_t)(row0 + lrow) * HH + kb + 16);
    bb.q[0] = *(const uint4*)(W2T + (size_t)(n0 + lrow) * HH + kb);
    bb.q[1] = *(const uint4*)(W2T + (size_t)(n0 + lrow) * HH + kb + 16);
    acc = __builtin_amdgcn_wmma_f32_16x16x32_bf16(false, a.v, false, bb.v,
                                                  (short)0, acc, false, false);
  }
  const int cn = n0 + lrow;
  const int rbase = row0 + ((lane >> 4) << 3);
  #pragma unroll
  for (int r = 0; r < 8; ++r)
    out[(size_t)(rbase + r) * VV + cn] = acc[r] + b2[cn];
}

// ---------------------------------------------------------------------------
extern "C" void kernel_launch(void* const* d_in, const int* in_sizes, int n_in,
                              void* d_out, int out_size, void* d_ws, size_t ws_size,
                              hipStream_t stream) {
  const float* x      = (const float*)d_in[0];
  const float* memory = (const float*)d_in[1];
  const int*   mtim   = (const int*)d_in[2];
  const float* msur   = (const float*)d_in[3];
  const float* lastp  = (const float*)d_in[4];
  const float* W1     = (const float*)d_in[5];
  const float* b1     = (const float*)d_in[6];
  const float* Wg     = (const float*)d_in[7];
  const float* bg     = (const float*)d_in[8];
  const float* W2     = (const float*)d_in[9];
  const float* b2     = (const float*)d_in[10];
  float* out = (float*)d_out;

  char* ws = (char*)d_ws;
  unsigned short* Abf     = (unsigned short*)ws;                        // 18,350,080 B
  float*          partial = (float*)(ws + 18350080);                    //  8,388,608 B
  unsigned short* hbf     = (unsigned short*)(ws + 18350080 + 8388608); //    524,288 B
  unsigned short* W2T     = (unsigned short*)(ws + 18350080 + 8388608 + 524288);

  prep_kernel<<<BB, 256, 0, stream>>>(x, memory, mtim, msur, lastp, Abf);
  w2t_kernel<<<(HH * VV) / 256, 256, 0, stream>>>(W2, W2T);
  gemm1_kernel<<<dim3(HH / 16, 2, KSPLIT), 256, SMEM_BYTES, stream>>>(Abf, W1, Wg, partial);
  gate_kernel<<<(BB * HH) / 256, 256, 0, stream>>>(partial, b1, bg, hbf);
  gemm2_kernel<<<BB / 16, 256, 0, stream>>>(hbf, W2T, b2, out);
}
